// GAT_16252156248491
// MI455X (gfx1250) — compile-verified, matrix operand order checked by reading, not run on hardware
//
#include <hip/hip_runtime.h>

// ---------------- problem constants (match reference) ----------------
constexpr int NN   = 50000;   // nodes
constexpr int NE   = 800000;  // edges
constexpr int IND  = 128;     // IN
constexpr int HIDD = 64;      // HID
constexpr int OUTD = 64;      // OUT
constexpr int HDS  = 2;       // heads
constexpr float NEG_SLOPE = 0.2f;

typedef __attribute__((ext_vector_type(16))) _Float16 v16h;
typedef __attribute__((ext_vector_type(8)))  float    v8f;

// =====================================================================
// WMMA fragment k-mapping (16-bit A 16x32 / symmetric B 32x16):
//   lane L, element j (0..15) -> k = 16*(j>>3) + 8*(L>>4) + (j&7)
// C/D: VGPR r -> row M = 8*(L>>4)+r, col N = L%16.
// =====================================================================

// ---------------------------------------------------------------------
// Repack B (f32, [K x Nc] row-major) into f16 fragment order:
//   Bp[((tileN*nkc + kc)*32 + lane)*16 + j]  = f16(B[k(j,lane)][tileN*16+lane%16])
// One wave per (tileN, kc); <= 32 waves total (weights are tiny).
// ---------------------------------------------------------------------
__global__ void repack_B_kernel(const float* __restrict__ B,
                                _Float16* __restrict__ Bp,
                                int Nc, int K)
{
    const int nkc  = K >> 5;
    const int wid  = blockIdx.x * (blockDim.x >> 5) + (threadIdx.x >> 5);
    const int lane = threadIdx.x & 31;
    const int total = (Nc >> 4) * nkc;
    if (wid >= total) return;
    const int tileN = wid / nkc;
    const int kc    = wid - tileN * nkc;
    const int ncol  = (tileN << 4) + (lane & 15);
    const int kb    = (kc << 5) + ((lane >> 4) << 3);
    _Float16* dst = Bp + ((size_t)wid * 32 + lane) * 16;
    #pragma unroll
    for (int j = 0; j < 8; ++j) {
        dst[j]     = (_Float16)B[(size_t)(kb + j) * Nc + ncol];
        dst[j + 8] = (_Float16)B[(size_t)(kb + 16 + j) * Nc + ncol];
    }
}

// ---------------------------------------------------------------------
// Panel GEMM: one wave computes a 16 x (TN*16) output panel.
// A: fp32 [M x K] row-major (converted to f16 in-register).
// Bp: f16 fragment-packed (from repack_B_kernel).
// TN unrolled -> TN back-to-back v_wmma per k-chunk, A fragment reused.
// ---------------------------------------------------------------------
template<int TN>
__global__ void gemm_wmma_panel_kernel(const float* __restrict__ A,
                                       const _Float16* __restrict__ Bp,
                                       const float* __restrict__ bias,
                                       float* __restrict__ C,
                                       int M, int K, int relu)
{
    constexpr int Nc = TN * 16;
    const int lane = threadIdx.x & 31;
    const int wid  = blockIdx.x * (blockDim.x >> 5) + (threadIdx.x >> 5);
    if (wid >= (M >> 4)) return;            // wave-uniform guard (EXEC all-1s inside)
    const int nkc  = K >> 5;
    const int mrow = (wid << 4) + (lane & 15);
    const int koff = (lane >> 4) << 3;      // 0 or 8

    v8f zero = {};
    v8f acc[TN];
    #pragma unroll
    for (int t = 0; t < TN; ++t) acc[t] = zero;

    for (int kc = 0; kc < nkc; ++kc) {
        // ---- A fragment: two contiguous 8-float runs -> 4x float4 loads
        const float* Ap = A + (size_t)mrow * K + (kc << 5) + koff;
        if (kc + 1 < nkc) __builtin_prefetch(Ap + 32, 0, 3);   // global_prefetch next chunk
        float4 f0 = *(const float4*)(Ap);
        float4 f1 = *(const float4*)(Ap + 4);
        float4 f2 = *(const float4*)(Ap + 16);
        float4 f3 = *(const float4*)(Ap + 20);
        v16h a;
        a[0]=(_Float16)f0.x; a[1]=(_Float16)f0.y; a[2]=(_Float16)f0.z; a[3]=(_Float16)f0.w;
        a[4]=(_Float16)f1.x; a[5]=(_Float16)f1.y; a[6]=(_Float16)f1.z; a[7]=(_Float16)f1.w;
        a[8]=(_Float16)f2.x; a[9]=(_Float16)f2.y; a[10]=(_Float16)f2.z; a[11]=(_Float16)f2.w;
        a[12]=(_Float16)f3.x; a[13]=(_Float16)f3.y; a[14]=(_Float16)f3.z; a[15]=(_Float16)f3.w;

        #pragma unroll
        for (int t = 0; t < TN; ++t) {
            // 32 B contiguous per lane, fully coalesced
            v16h b = *(const v16h*)(Bp + ((size_t)(t * nkc + kc) * 32 + lane) * 16);
            acc[t] = __builtin_amdgcn_wmma_f32_16x16x32_f16(
                         false, a, false, b, (short)0, acc[t], false, false);
        }
    }

    const int mbase = (wid << 4) + ((lane >> 4) << 3);
    #pragma unroll
    for (int t = 0; t < TN; ++t) {
        const int ncol = (t << 4) + (lane & 15);
        const float bv = bias ? bias[ncol] : 0.0f;
        #pragma unroll
        for (int r = 0; r < 8; ++r) {
            float v = acc[t][r] + bv;
            if (relu) v = fmaxf(v, 0.0f);
            C[(size_t)(mbase + r) * Nc + ncol] = v;
        }
    }
}

// =====================================================================
// Attention logits: el[n,h] = <feat[n,h,:], al[h,:]>, er likewise.
// =====================================================================
__global__ void elr_kernel(const float* __restrict__ feat,
                           const float* __restrict__ al,
                           const float* __restrict__ ar,
                           float* __restrict__ el,
                           float* __restrict__ er)
{
    const int wid  = blockIdx.x * (blockDim.x >> 5) + (threadIdx.x >> 5);
    const int lane = threadIdx.x & 31;
    if (wid >= NN) return;

    float4 f  = ((const float4*)(feat + (size_t)wid * 128))[lane];
    float4 a4 = ((const float4*)al)[lane];
    float4 r4 = ((const float4*)ar)[lane];
    float pl = f.x * a4.x + f.y * a4.y + f.z * a4.z + f.w * a4.w;
    float pr = f.x * r4.x + f.y * r4.y + f.z * r4.z + f.w * r4.w;
    #pragma unroll
    for (int off = 1; off < 16; off <<= 1) {
        pl += __shfl_xor(pl, off, 32);
        pr += __shfl_xor(pr, off, 32);
    }
    if ((lane & 15) == 0) {
        int h = lane >> 4;
        el[wid * HDS + h] = pl;
        er[wid * HDS + h] = pr;
    }
}

// ---------------- init kernels ----------------
__global__ void init_mz_kernel(float* __restrict__ m, float* __restrict__ z)
{
    int i = blockIdx.x * blockDim.x + threadIdx.x;
    if (i < NN * HDS) { m[i] = __int_as_float(0xff800000); z[i] = 0.0f; }
}

__global__ void init_bias_kernel(float* __restrict__ agg,
                                 const float* __restrict__ bias)
{
    size_t i = (size_t)blockIdx.x * blockDim.x + threadIdx.x;
    if (i < (size_t)NN * 128) agg[i] = bias[i & 127];
}

// ---------------- edge kernels ----------------
__device__ __forceinline__ float lrelu(float v) {
    return v > 0.0f ? v : NEG_SLOPE * v;
}

__device__ __forceinline__ void atomicMaxFloat(float* addr, float val)
{
    int iv = __float_as_int(val);
    if (iv >= 0) atomicMax((int*)addr, iv);
    else         atomicMin((unsigned int*)addr, (unsigned int)iv);
}

__global__ void edge_max_kernel(const int* __restrict__ src,
                                const int* __restrict__ dst,
                                const float* __restrict__ el,
                                const float* __restrict__ er,
                                float* __restrict__ mbuf)
{
    int e = blockIdx.x * blockDim.x + threadIdx.x;
    if (e >= NE) return;
    int s = src[e], d = dst[e];
    #pragma unroll
    for (int h = 0; h < HDS; ++h) {
        float v = lrelu(el[s * HDS + h] + er[d * HDS + h]);
        atomicMaxFloat(&mbuf[d * HDS + h], v);
    }
}

__global__ void edge_expsum_kernel(const int* __restrict__ src,
                                   const int* __restrict__ dst,
                                   const float* __restrict__ el,
                                   const float* __restrict__ er,
                                   const float* __restrict__ mbuf,
                                   float* __restrict__ zbuf,
                                   float* __restrict__ exb)
{
    int e = blockIdx.x * blockDim.x + threadIdx.x;
    if (e >= NE) return;
    int s = src[e], d = dst[e];
    #pragma unroll
    for (int h = 0; h < HDS; ++h) {
        float v = lrelu(el[s * HDS + h] + er[d * HDS + h]);
        float x = expf(v - mbuf[d * HDS + h]);
        exb[e * HDS + h] = x;
        atomicAdd(&zbuf[d * HDS + h], x);
    }
}

// wave per edge: agg[dst, :] += (ex/z) * feat[src, :]   (128 comps, 4/lane)
__global__ void edge_agg_kernel(const int* __restrict__ src,
                                const int* __restrict__ dst,
                                const float* __restrict__ feat,
                                const float* __restrict__ exb,
                                const float* __restrict__ zbuf,
                                float* __restrict__ agg)
{
    const int wid  = blockIdx.x * (blockDim.x >> 5) + (threadIdx.x >> 5);
    const int lane = threadIdx.x & 31;
    if (wid >= NE) return;
    const int s = src[wid], d = dst[wid];
    const int h = lane >> 4;                       // comps 4l..4l+3 all in head l/16
    const float coef = exb[wid * HDS + h] / zbuf[d * HDS + h];
    float4 f = ((const float4*)(feat + (size_t)s * 128))[lane];
    float* op = agg + (size_t)d * 128 + lane * 4;
    atomicAdd(op + 0, coef * f.x);
    atomicAdd(op + 1, coef * f.y);
    atomicAdd(op + 2, coef * f.z);
    atomicAdd(op + 3, coef * f.w);
}

// =====================================================================
extern "C" void kernel_launch(void* const* d_in, const int* in_sizes, int n_in,
                              void* d_out, int out_size, void* d_ws, size_t ws_size,
                              hipStream_t stream)
{
    const float* h      = (const float*)d_in[0];
    const int*   src    = (const int*)  d_in[1];
    const int*   dst    = (const int*)  d_in[2];
    const float* W1     = (const float*)d_in[3];
    const float* al1    = (const float*)d_in[4];
    const float* ar1    = (const float*)d_in[5];
    const float* b1     = (const float*)d_in[6];
    const float* lin1W  = (const float*)d_in[7];
    const float* lin1b  = (const float*)d_in[8];
    const float* W2     = (const float*)d_in[9];
    const float* al2    = (const float*)d_in[10];
    const float* ar2    = (const float*)d_in[11];
    const float* b2     = (const float*)d_in[12];
    const float* lin2W  = (const float*)d_in[13];
    const float* lin2b  = (const float*)d_in[14];
    float* out = (float*)d_out;

    // ---- workspace layout (floats) ----
    float* ws   = (float*)d_ws;
    float* feat = ws;                               // NN*128
    float* agg  = feat + (size_t)NN * 128;          // NN*128
    float* xbuf = agg  + (size_t)NN * 128;          // NN*64
    float* el   = xbuf + (size_t)NN * 64;           // NN*2
    float* er   = el   + (size_t)NN * HDS;          // NN*2
    float* mbuf = er   + (size_t)NN * HDS;          // NN*2
    float* zbuf = mbuf + (size_t)NN * HDS;          // NN*2
    float* exb  = zbuf + (size_t)NN * HDS;          // NE*2
    _Float16* bpack = (_Float16*)(exb + (size_t)NE * HDS);  // <= 128*128 f16 (32B-aligned)

    auto gemm = [&](const float* A, const float* Bsrc, const float* bias,
                    float* C, int M, int Nc, int K, int relu) {
        const int nkc = K / 32;
        const int packWaves = (Nc / 16) * nkc;
        repack_B_kernel<<<(packWaves + 7) / 8, 256, 0, stream>>>(Bsrc, bpack, Nc, K);
        const int blocks = (M / 16 + 7) / 8;        // wave per 16-row panel
        if (Nc == 128)
            gemm_wmma_panel_kernel<8><<<blocks, 256, 0, stream>>>(A, bpack, bias, C, M, K, relu);
        else
            gemm_wmma_panel_kernel<4><<<blocks, 256, 0, stream>>>(A, bpack, bias, C, M, K, relu);
    };

    const int nodeWaves = (NN + 7) / 8;
    const int edgeThr   = (NE + 255) / 256;
    const int edgeWaves = (NE + 7) / 8;
    const int mzBlocks  = (NN * HDS + 255) / 256;
    const int aggBlocks = (int)(((size_t)NN * 128 + 255) / 256);

    // ================= layer 1: GATConv(IN -> HID, H=2) =================
    gemm(h, W1, nullptr, feat, NN, HDS * HIDD, IND, 0);             // feat1 = h @ W1
    elr_kernel<<<nodeWaves, 256, 0, stream>>>(feat, al1, ar1, el, er);
    init_mz_kernel<<<mzBlocks, 256, 0, stream>>>(mbuf, zbuf);
    init_bias_kernel<<<aggBlocks, 256, 0, stream>>>(agg, b1);       // agg = b1 (bias)
    edge_max_kernel   <<<edgeThr, 256, 0, stream>>>(src, dst, el, er, mbuf);
    edge_expsum_kernel<<<edgeThr, 256, 0, stream>>>(src, dst, el, er, mbuf, zbuf, exb);
    edge_agg_kernel   <<<edgeWaves, 256, 0, stream>>>(src, dst, feat, exb, zbuf, agg);
    gemm(agg, lin1W, lin1b, xbuf, NN, HIDD, HDS * HIDD, 1);         // x = relu(agg@lin1W+b)

    // ================= layer 2: GATConv(HID -> OUT, H=2) =================
    gemm(xbuf, W2, nullptr, feat, NN, HDS * OUTD, HIDD, 0);         // feat2 = x @ W2
    elr_kernel<<<nodeWaves, 256, 0, stream>>>(feat, al2, ar2, el, er);
    init_mz_kernel<<<mzBlocks, 256, 0, stream>>>(mbuf, zbuf);
    init_bias_kernel<<<aggBlocks, 256, 0, stream>>>(agg, b2);
    edge_max_kernel   <<<edgeThr, 256, 0, stream>>>(src, dst, el, er, mbuf);
    edge_expsum_kernel<<<edgeThr, 256, 0, stream>>>(src, dst, el, er, mbuf, zbuf, exb);
    edge_agg_kernel   <<<edgeWaves, 256, 0, stream>>>(src, dst, feat, exb, zbuf, agg);
    gemm(agg, lin2W, lin2b, out, NN, OUTD, HDS * OUTD, 0);          // out = agg@lin2W+b
}
